// ESMMultiHeadAttention_30442728194638
// MI455X (gfx1250) — matrix operand order repaired and checked by reading, not compiled, not run on hardware
//
#include <hip/hip_runtime.h>

#define S      2048
#define DM     2048
#define NH     32
#define DH     64
#define N3     6144
#define EPSF   1e-5f
#define NEGINF (-__builtin_inff())

typedef __attribute__((ext_vector_type(16))) _Float16 v16h;
typedef __attribute__((ext_vector_type(8)))  float    v8f;
typedef __attribute__((ext_vector_type(4)))  unsigned int v4u;

static __device__ __forceinline__ v8f wmma16(v16h a, v16h b, v8f c) {
    return __builtin_amdgcn_wmma_f32_16x16x32_f16(false, a, false, b, (short)0, c, false, false);
}

// A-matrix fragment (16xK tile, K-step 32), row-major source, K contiguous.
// ISA layout: lanes 0-15 hold row M=lane, K {k0..k0+7, k0+16..k0+23};
// lanes 16-31 hold row M=lane-16, K {k0+8..k0+15, k0+24..k0+31}.
static __device__ __forceinline__ v16h load_fragA(const _Float16* base, int ld, int row0, int k0) {
    int lane = threadIdx.x & 31;
    const _Float16* p = base + (size_t)(row0 + (lane & 15)) * ld + k0 + ((lane >> 4) << 3);
    union { v4u u[2]; v16h h; } f;
    f.u[0] = *(const v4u*)p;
    f.u[1] = *(const v4u*)(p + 16);
    return f.h;
}

// B-matrix fragment: source row-major [N][K] (B transposed, K contiguous).
// ISA layout: lanes 0-15 hold col N=lane, K {k0..k0+7, k0+16..k0+23};
// lanes 16-31 hold K {k0+8..k0+15, k0+24..k0+31}.
static __device__ __forceinline__ v16h load_fragB(const _Float16* base, int ld, int col0, int k0) {
    int lane = threadIdx.x & 31;
    const _Float16* p = base + (size_t)(col0 + (lane & 15)) * ld + k0 + ((lane >> 4) << 3);
    union { v4u u[2]; v16h h; } f;
    f.u[0] = *(const v4u*)p;
    f.u[1] = *(const v4u*)(p + 16);
    return f.h;
}

// block-wide sum over 256 threads; result broadcast to all.
static __device__ __forceinline__ float blk_sum(float v, float* tmp) {
    #pragma unroll
    for (int off = 16; off; off >>= 1) v += __shfl_xor(v, off, 32);
    __syncthreads();
    if ((threadIdx.x & 31) == 0) tmp[threadIdx.x >> 5] = v;
    __syncthreads();
    float s = 0.f;
    #pragma unroll
    for (int i = 0; i < 8; i++) s += tmp[i];
    return s;
}

// ---------------- kernel 1: LayerNorm(x) -> f16 ----------------
__global__ void ln_x_kernel(const float* __restrict__ x, const float* __restrict__ w,
                            const float* __restrict__ b, _Float16* __restrict__ h16) {
    __shared__ float tmp[8];
    int s = blockIdx.x;
    const float* row = x + (size_t)s * DM;
    float sum = 0.f, sq = 0.f;
    for (int i = threadIdx.x; i < DM; i += blockDim.x) {
        float v = row[i]; sum += v; sq += v * v;
    }
    float mu  = blk_sum(sum, tmp) * (1.0f / DM);
    float ms  = blk_sum(sq,  tmp) * (1.0f / DM);
    float rstd = rsqrtf(ms - mu * mu + EPSF);
    _Float16* out = h16 + (size_t)s * DM;
    for (int i = threadIdx.x; i < DM; i += blockDim.x)
        out[i] = (_Float16)((row[i] - mu) * rstd * w[i] + b[i]);
}

// ---------------- kernel 2: fp32 -> f16 convert ----------------
__global__ void cvt_kernel(const float* __restrict__ in, _Float16* __restrict__ out, int n) {
    int i = blockIdx.x * blockDim.x + threadIdx.x;
    int stride = gridDim.x * blockDim.x;
    for (; i < n; i += stride) out[i] = (_Float16)in[i];
}

// ---------------- kernels 3/6: LDS-tiled WMMA GEMM  C[M,N] = A[M,K] * B[N,K]^T ----
// block = 8 waves computes 128(M) x 128(N); wave = 32x64; BK = 32.
#define BM 128
#define BN 128
#define BK 32
__global__ void gemm16_kernel(const _Float16* __restrict__ A, const _Float16* __restrict__ B,
                              float* __restrict__ C, int M, int N, int K, int tilesN) {
    __shared__ __align__(16) _Float16 As[BM * BK];
    __shared__ __align__(16) _Float16 Bs[BN * BK];
    int tid = threadIdx.x;
    int waveId = tid >> 5;
    int wm = waveId >> 1, wn = waveId & 1;           // 4 x 2 wave grid
    int m0 = (blockIdx.x / tilesN) * BM;
    int n0 = (blockIdx.x % tilesN) * BN;
    v8f c[2][4] = {};
    for (int k0 = 0; k0 < K; k0 += BK) {
        __syncthreads();
        #pragma unroll
        for (int i = 0; i < 2; i++) {                // 512 16B-chunks, 2 per thread
            int ch = tid + i * 256;
            int row = ch >> 2, off = (ch & 3) << 3;
            *(v4u*)(As + row * BK + off) = *(const v4u*)(A + (size_t)(m0 + row) * K + k0 + off);
            *(v4u*)(Bs + row * BK + off) = *(const v4u*)(B + (size_t)(n0 + row) * K + k0 + off);
            if (k0 + BK < K) {                       // global_prefetch_b8 of next k-tile
                __builtin_prefetch(A + (size_t)(m0 + row) * K + k0 + BK + off, 0, 3);
                __builtin_prefetch(B + (size_t)(n0 + row) * K + k0 + BK + off, 0, 3);
            }
        }
        __syncthreads();
        v16h a0 = load_fragA(As, BK, wm * 32,      0);
        v16h a1 = load_fragA(As, BK, wm * 32 + 16, 0);
        #pragma unroll
        for (int t = 0; t < 4; t++) {
            v16h b = load_fragB(Bs, BK, wn * 64 + t * 16, 0);
            c[0][t] = wmma16(a0, b, c[0][t]);
            c[1][t] = wmma16(a1, b, c[1][t]);
        }
    }
    int lane = tid & 31, col = lane & 15, hl = lane >> 4;
    #pragma unroll
    for (int mt = 0; mt < 2; mt++) {
        #pragma unroll
        for (int j = 0; j < 8; j++) {
            int r = m0 + wm * 32 + mt * 16 + j + 8 * hl;
            float* cr = C + (size_t)r * N + n0 + wn * 64 + col;
            cr[0] = c[mt][0][j]; cr[16] = c[mt][1][j];
            cr[32] = c[mt][2][j]; cr[48] = c[mt][3][j];
        }
    }
}

// ---------------- kernel 4: q/k LayerNorm + RoPE, v transpose ----------------
__global__ void qk_prep_kernel(const float* __restrict__ qkv,
                               const float* __restrict__ qw, const float* __restrict__ kw,
                               _Float16* __restrict__ q16, _Float16* __restrict__ k16,
                               _Float16* __restrict__ vT16) {
    __shared__ float tmp[8];
    int s = blockIdx.x;
    const float* qq = qkv + (size_t)s * N3;
    const float* kk = qq + DM;
    const float* vv = qq + 2 * DM;
    float sq = 0.f, sq2 = 0.f, sk = 0.f, sk2 = 0.f;
    for (int i = threadIdx.x; i < DM; i += blockDim.x) {
        float a = qq[i]; sq += a; sq2 += a * a;
        float b = kk[i]; sk += b; sk2 += b * b;
    }
    float muq = blk_sum(sq,  tmp) * (1.0f / DM);
    float mq2 = blk_sum(sq2, tmp) * (1.0f / DM);
    float muk = blk_sum(sk,  tmp) * (1.0f / DM);
    float mk2 = blk_sum(sk2, tmp) * (1.0f / DM);
    float rq = rsqrtf(mq2 - muq * muq + EPSF);
    float rk = rsqrtf(mk2 - muk * muk + EPSF);
    const float LTH = 0.28782313662425576f;  // ln(10000)/32
    for (int i = threadIdx.x; i < DM; i += blockDim.x) {
        int h = i >> 6, d = i & 63;
        int fi = (d < 32) ? d : d - 32;
        int ip = (d < 32) ? i + 32 : i - 32;
        float th = (float)s * __expf(-(float)fi * LTH);
        float c = cosf(th), sn = sinf(th);
        float qn  = (qq[i]  - muq) * rq * qw[i];
        float qnp = (qq[ip] - muq) * rq * qw[ip];
        float kn  = (kk[i]  - muk) * rk * kw[i];
        float knp = (kk[ip] - muk) * rk * kw[ip];
        float qo = (d < 32) ? (qn * c - qnp * sn) : (qn * c + qnp * sn);
        float ko = (d < 32) ? (kn * c - knp * sn) : (kn * c + knp * sn);
        size_t hi = ((size_t)h * S + s) * DH + d;
        q16[hi] = (_Float16)qo;
        k16[hi] = (_Float16)ko;
        vT16[((size_t)h * DH + d) * S + s] = (_Float16)vv[i];
    }
}

// ---------------- kernel 5: masked flash attention, block-shared K/V ----------
// block = 8 waves, one head, 128 consecutive query rows (16 per wave).
// per 32-key chunk: K (32x64) and V (64x32) staged once into LDS for all waves.
__global__ void attn_kernel(const _Float16* __restrict__ q16, const _Float16* __restrict__ k16,
                            const _Float16* __restrict__ vT16, const int* __restrict__ seq,
                            _Float16* __restrict__ ctx16) {
    __shared__ __align__(16) _Float16 Ksh[32 * 64];
    __shared__ __align__(16) _Float16 Vsh[64 * 32];
    __shared__ __align__(16) _Float16 pbuf[8 * 16 * 32];
    int tid = threadIdx.x;
    int waveId = tid >> 5, lane = tid & 31, col = lane & 15, hl = lane >> 4;
    int h  = blockIdx.x >> 4;                 // 16 blocks per head
    int s0 = (blockIdx.x & 15) * 128 + waveId * 16;
    _Float16* pt = pbuf + waveId * 512;

    const _Float16* Qh = q16  + (size_t)h * S * DH;
    const _Float16* Kh = k16  + (size_t)h * S * DH;
    const _Float16* Vh = vT16 + (size_t)h * DH * S;

    v16h aq0 = load_fragA(Qh, DH, s0, 0);
    v16h aq1 = load_fragA(Qh, DH, s0, 32);

    int srow[8];
    #pragma unroll
    for (int j = 0; j < 8; j++) srow[j] = seq[s0 + j + 8 * hl];

    float m[8], ls[8];
    v8f acc0 = {}, acc1 = {}, acc2 = {}, acc3 = {};
    #pragma unroll
    for (int j = 0; j < 8; j++) { m[j] = NEGINF; ls[j] = 0.f; }

    const float scale = 0.125f;  // 1/sqrt(64)
    for (int t0 = 0; t0 < S; t0 += 32) {
        __syncthreads();   // all waves done reading previous K/V chunk
        {   // K chunk: 32 rows x 64 halves = 256 x 16B, one per thread
            int row = tid >> 3, off = (tid & 7) << 3;
            *(v4u*)(Ksh + row * 64 + off) = *(const v4u*)(Kh + (size_t)(t0 + row) * DH + off);
        }
        {   // V chunk: 64 rows x 32 halves = 256 x 16B, one per thread
            int row = tid >> 2, off = (tid & 3) << 3;
            *(v4u*)(Vsh + row * 32 + off) = *(const v4u*)(Vh + (size_t)row * S + t0 + off);
        }
        __syncthreads();

        v8f l0 = {}, l1 = {};
        l0 = wmma16(aq0, load_fragB(Ksh, DH, 0,  0),  l0);
        l0 = wmma16(aq1, load_fragB(Ksh, DH, 0,  32), l0);
        l1 = wmma16(aq0, load_fragB(Ksh, DH, 16, 0),  l1);
        l1 = wmma16(aq1, load_fragB(Ksh, DH, 16, 32), l1);

        int sc0 = seq[t0 + col], sc1 = seq[t0 + 16 + col];
        #pragma unroll
        for (int j = 0; j < 8; j++) {
            bool k0m = (srow[j] == sc0), k1m = (srow[j] == sc1);
            float a0 = l0[j] * scale, a1 = l1[j] * scale;
            float mx = fmaxf(k0m ? a0 : NEGINF, k1m ? a1 : NEGINF);
            mx = fmaxf(mx, __shfl_xor(mx, 1, 32));
            mx = fmaxf(mx, __shfl_xor(mx, 2, 32));
            mx = fmaxf(mx, __shfl_xor(mx, 4, 32));
            mx = fmaxf(mx, __shfl_xor(mx, 8, 32));
            float nm = fmaxf(m[j], mx);
            float cf = (m[j] == nm) ? 1.0f : __expf(m[j] - nm);
            float e0 = k0m ? __expf(a0 - nm) : 0.0f;
            float e1 = k1m ? __expf(a1 - nm) : 0.0f;
            float rs = e0 + e1;
            rs += __shfl_xor(rs, 1, 32);
            rs += __shfl_xor(rs, 2, 32);
            rs += __shfl_xor(rs, 4, 32);
            rs += __shfl_xor(rs, 8, 32);
            ls[j] = ls[j] * cf + rs;
            m[j] = nm;
            acc0[j] *= cf; acc1[j] *= cf; acc2[j] *= cf; acc3[j] *= cf;
            int r = (j + 8 * hl) * 32;
            pt[r + col]      = (_Float16)e0;
            pt[r + 16 + col] = (_Float16)e1;
        }
        __builtin_amdgcn_wave_barrier();        // order ds stores before reload
        v16h pa = load_fragA(pt, 32, 0, 0);     // P tile in A-fragment layout
        acc0 = wmma16(pa, load_fragB(Vsh, 32, 0,  0), acc0);
        acc1 = wmma16(pa, load_fragB(Vsh, 32, 16, 0), acc1);
        acc2 = wmma16(pa, load_fragB(Vsh, 32, 32, 0), acc2);
        acc3 = wmma16(pa, load_fragB(Vsh, 32, 48, 0), acc3);
        __builtin_amdgcn_wave_barrier();
    }
    #pragma unroll
    for (int j = 0; j < 8; j++) {
        float inv = 1.0f / ls[j];
        size_t base = (size_t)(s0 + j + 8 * hl) * DM + h * DH + col;
        ctx16[base]      = (_Float16)(acc0[j] * inv);
        ctx16[base + 16] = (_Float16)(acc1[j] * inv);
        ctx16[base + 32] = (_Float16)(acc2[j] * inv);
        ctx16[base + 48] = (_Float16)(acc3[j] * inv);
    }
}

extern "C" void kernel_launch(void* const* d_in, const int* in_sizes, int n_in,
                              void* d_out, int out_size, void* d_ws, size_t ws_size,
                              hipStream_t stream) {
    (void)in_sizes; (void)n_in; (void)out_size; (void)ws_size;
    const float* x      = (const float*)d_in[0];
    const float* ln_w   = (const float*)d_in[1];
    const float* ln_b   = (const float*)d_in[2];
    const float* Wqkv   = (const float*)d_in[3];
    const float* q_ln_w = (const float*)d_in[4];
    const float* k_ln_w = (const float*)d_in[5];
    const float* Wout   = (const float*)d_in[6];
    const int*   seq    = (const int*)d_in[7];

    char* ws = (char*)d_ws;
    const size_t MB = 1024ull * 1024ull;
    _Float16* h16     = (_Float16*)(ws);              //  8 MB
    _Float16* Wqkv16  = (_Float16*)(ws + 8   * MB);   // 24 MB
    _Float16* Wout16  = (_Float16*)(ws + 32  * MB);   //  8 MB
    float*    qkv     = (float*)   (ws + 40  * MB);   // 48 MB
    _Float16* q16     = (_Float16*)(ws + 88  * MB);   //  8 MB
    _Float16* k16     = (_Float16*)(ws + 96  * MB);   //  8 MB
    _Float16* vT16    = (_Float16*)(ws + 104 * MB);   //  8 MB
    _Float16* ctx16   = (_Float16*)(ws + 112 * MB);   //  8 MB

    ln_x_kernel<<<S, 256, 0, stream>>>(x, ln_w, ln_b, h16);
    cvt_kernel<<<4096, 256, 0, stream>>>(Wqkv, Wqkv16, 3 * DM * DM);
    cvt_kernel<<<2048, 256, 0, stream>>>(Wout, Wout16, DM * DM);
    // qkv = h @ Wqkv.T : M=2048, N=6144, K=2048 ; (16 x 48) 128x128 block tiles
    gemm16_kernel<<<16 * 48, 256, 0, stream>>>(h16, Wqkv16, qkv, S, N3, DM, N3 / BN);
    qk_prep_kernel<<<S, 256, 0, stream>>>(qkv, q_ln_w, k_ln_w, q16, k16, vT16);
    // attention: 32 heads x 16 blocks (128 q-rows each)
    attn_kernel<<<NH * 16, 256, 0, stream>>>(q16, k16, vT16, seq, ctx16);
    // out = ctx @ Wout.T : M=2048, N=2048, K=2048 ; (16 x 16) block tiles
    gemm16_kernel<<<16 * 16, 256, 0, stream>>>(ctx16, Wout16, (float*)d_out, S, DM, DM, DM / BN);
}
